// SlotSetEncoder_3736621548316
// MI455X (gfx1250) — matrix operand order repaired and checked by reading, not compiled, not run on hardware
//
#include <hip/hip_runtime.h>
#include <math.h>

// ---------------------------------------------------------------------------
// SlotSetEncoder fused for MI455X (gfx1250, wave32, WMMA f32_16x16x32_f16)
//
// Pipeline (reads X exactly once, ~134 MB -> ~6us HBM floor):
//   A1: S = LayerNorm(slots_w)                       (64x512)
//   A2: Qp[h] = (S_h @ Wk_h) * 1/sqrt(hd) -> f16;  qb[h,k] = (S_h[k]·bk_h)/sqrt(hd)
//   A3: Wv -> f16
//   B : per (b, 128-token chunk): stage X chunk f16 in LDS; per head-wave:
//         L  = Qp_h @ Xc^T + qb          (WMMA, K-proj folded into queries)
//         Wn = colsoftmax_over_slots(L)  (shfl_xor(16) lane-pair reduction)
//         Vc = Xc @ Wv_h^T + bv          (WMMA)
//         U += Wn @ Vc ; Ck += rowsum(Wn)  -> global f32 atomics
//   C : out = LayerNorm( U/(Ck+eps) )    (B*K rows of 512)
// ---------------------------------------------------------------------------

typedef __attribute__((ext_vector_type(16))) _Float16 v16h;
typedef __attribute__((ext_vector_type(8)))  _Float16 v8h;
typedef __attribute__((ext_vector_type(8)))  float    v8f;

#define NB    8      // batch
#define NT    8192   // tokens
#define ND    512    // model dim
#define NH    8      // heads
#define HD    64     // head dim
#define KSL   64     // slots
#define TC    128    // tokens per block chunk
#define XSTR  520    // LDS row stride (halfs) for staged X chunk (pad vs banks)
#define WSTR  40     // LDS row stride (halfs) for per-wave 32-wide scratch
#define SQRTD 0.125f // 1/sqrt(64)
#define EPSW  1e-20f
#define LNEPS 1e-5f

// ---- workspace layout (bytes) ----
#define S_OFF    0u
#define S_BYTES  (KSL*ND*4u)                 // 131072
#define WV_OFF   (S_OFF + S_BYTES)
#define WV_BYTES (ND*ND*2u)                  // 524288
#define QP_OFF   (WV_OFF + WV_BYTES)
#define QP_BYTES (NH*KSL*ND*2u)              // 524288
#define QB_OFF   (QP_OFF + QP_BYTES)
#define QB_BYTES (NH*KSL*4u)                 // 2048
#define U_OFF    (QB_OFF + QB_BYTES)
#define U_BYTES  (NB*NH*KSL*HD*4u)           // 1048576
#define CS_OFF   (U_OFF + U_BYTES)
#define CS_BYTES (NB*NH*KSL*4u)              // 16384

struct __align__(8) h4 { _Float16 v[4]; };

// A-matrix tile (M=16 x K=32, f16). Lanes 0-15 hold M=lane,
// K = 0..15; lanes 16-31 hold M=lane-16, K = 16..31.
__device__ __forceinline__ v16h load_tile_a(const _Float16* __restrict__ base,
                                            int row0, int k0, int stride, int lane) {
    const int m  = lane & 15;
    const int hi = (lane >> 4) & 1;
    const _Float16* p = base + (row0 + m) * stride + k0 + hi * 16;
    v8h lo = *(const v8h*)(p);
    v8h hh = *(const v8h*)(p + 8);
    v16h out;
#pragma unroll
    for (int i = 0; i < 8; ++i) { out[i] = lo[i]; out[i + 8] = hh[i]; }
    return out;
}

// B-matrix tile (K=32 x N=16, f16), source stored row-major as [N][K].
// Lanes 0-15 hold N=lane, K=0..15; lanes 16-31 hold N=lane-16, K=16..31.
__device__ __forceinline__ v16h load_tile_b(const _Float16* __restrict__ base,
                                            int n0, int k0, int stride, int lane) {
    const int nn = lane & 15;
    const int hi = (lane >> 4) & 1;
    const _Float16* p = base + (n0 + nn) * stride + k0 + hi * 16;
    v8h lo = *(const v8h*)(p);
    v8h hh = *(const v8h*)(p + 8);
    v16h out;
#pragma unroll
    for (int i = 0; i < 8; ++i) { out[i] = lo[i]; out[i + 8] = hh[i]; }
    return out;
}

__device__ __forceinline__ v8f wmma16(v16h a, v16h b, v8f c) {
    return __builtin_amdgcn_wmma_f32_16x16x32_f16(false, a, false, b,
                                                  (short)0, c, false, false);
}

__device__ __forceinline__ float block_sum256(float v, float* red) {
    const int t = threadIdx.x;
    red[t] = v;
    __syncthreads();
#pragma unroll
    for (int s = 128; s > 0; s >>= 1) {
        if (t < s) red[t] += red[t + s];
        __syncthreads();
    }
    float r = red[0];
    __syncthreads();
    return r;
}

// ---------------- A1: LayerNorm of slots -> S (fp32) ----------------
__global__ void sse_slot_ln(const float* __restrict__ slots,
                            const float* __restrict__ g,
                            const float* __restrict__ bt,
                            float* __restrict__ S) {
    __shared__ float red[256];
    const int k = blockIdx.x;                 // 64 slot rows
    const int t = threadIdx.x;
    const float x0 = slots[k * ND + t];
    const float x1 = slots[k * ND + t + 256];
    const float mu  = block_sum256(x0 + x1, red) * (1.0f / ND);
    const float d0 = x0 - mu, d1 = x1 - mu;
    const float var = block_sum256(d0 * d0 + d1 * d1, red) * (1.0f / ND);
    const float rs = rsqrtf(var + LNEPS);
    S[k * ND + t]       = d0 * rs * g[t]       + bt[t];
    S[k * ND + t + 256] = d1 * rs * g[t + 256] + bt[t + 256];
}

// ---------------- A2: Qp[h,k,:] = (S_h[k] @ Wk_h)/sqrt(hd), qb ----------------
__global__ void sse_qproj(const float* __restrict__ S,
                          const float* __restrict__ Wk,
                          const float* __restrict__ bk,
                          _Float16* __restrict__ Qp,
                          float* __restrict__ qb) {
    __shared__ float sS[HD];
    const int h = blockIdx.x >> 6;
    const int k = blockIdx.x & 63;
    if (threadIdx.x < HD) sS[threadIdx.x] = S[k * ND + h * HD + threadIdx.x];
    __syncthreads();
#pragma unroll
    for (int j = 0; j < 2; ++j) {
        const int i = threadIdx.x + j * 256;
        float acc = 0.f;
#pragma unroll 8
        for (int d = 0; d < HD; ++d) acc += sS[d] * Wk[(h * HD + d) * ND + i];
        Qp[((size_t)(h * KSL + k)) * ND + i] = (_Float16)(acc * SQRTD);
    }
    if (threadIdx.x == 0) {
        float a = 0.f;
        for (int d = 0; d < HD; ++d) a += sS[d] * bk[h * HD + d];
        qb[h * KSL + k] = a * SQRTD;
    }
}

// ---------------- A3: fp32 -> f16 ----------------
__global__ void sse_cvt16(const float* __restrict__ src, _Float16* __restrict__ dst, int n) {
    const int i = blockIdx.x * blockDim.x + threadIdx.x;
    if (i < n) dst[i] = (_Float16)src[i];
}

// ---------------- B: fused attention over one (batch, 128-token) chunk ----------------
__global__ void __launch_bounds__(256, 1)
sse_fused(const float* __restrict__ X,
          const _Float16* __restrict__ Qp,
          const float* __restrict__ qb,
          const _Float16* __restrict__ Wv16,
          const float* __restrict__ bv,
          float* __restrict__ U,
          float* __restrict__ Cs) {
    extern __shared__ char smem[];
    _Float16* Xs = (_Float16*)smem;                         // [TC][XSTR] f16
    const int wave = threadIdx.x >> 5;                      // 0..7 == head
    const int lane = threadIdx.x & 31;
    const int hi = (lane >> 4) & 1;
    const int nn = lane & 15;
    _Float16* wnS = (_Float16*)(smem + TC * XSTR * 2) + wave * (2 * KSL * WSTR);
    _Float16* vtS = wnS + KSL * WSTR;

    const int b     = blockIdx.x >> 6;
    const int chunk = blockIdx.x & 63;
    const int t0    = chunk * TC;

    // ---- stage 128x512 fp32 -> f16 LDS (each thread 64 float4 loads) ----
    for (int idx = threadIdx.x; idx < (TC * ND) / 4; idx += 256) {
        const int row = idx >> 7;               // 128 float4 per row
        const int c4  = (idx & 127) << 2;
        const float4 f = *(const float4*)(X + ((size_t)(b * NT + t0 + row)) * ND + c4);
        h4 hh;
        hh.v[0] = (_Float16)f.x; hh.v[1] = (_Float16)f.y;
        hh.v[2] = (_Float16)f.z; hh.v[3] = (_Float16)f.w;
        *(h4*)(Xs + row * XSTR + c4) = hh;
    }
    __syncthreads();

    const int h = wave;
    const _Float16* QpH = Qp + (size_t)h * KSL * ND;
    const _Float16* WvH = Wv16 + (size_t)h * HD * ND;

    // per-lane constants: qb for my C-layout rows, bv for my C-layout cols
    float qbr[4][8];
#pragma unroll
    for (int rb = 0; rb < 4; ++rb)
#pragma unroll
        for (int r = 0; r < 8; ++r)
            qbr[rb][r] = qb[h * KSL + rb * 16 + r + 8 * hi];
    float bvr[4];
#pragma unroll
    for (int cb = 0; cb < 4; ++cb) bvr[cb] = bv[h * HD + cb * 16 + nn];

    v8f uacc[16];                      // U tiles [slot-rb][d-cb]
#pragma unroll
    for (int t = 0; t < 16; ++t)
#pragma unroll
        for (int r = 0; r < 8; ++r) uacc[t][r] = 0.f;
    float cacc[4][8];                  // row-sum partials
#pragma unroll
    for (int rb = 0; rb < 4; ++rb)
#pragma unroll
        for (int r = 0; r < 8; ++r) cacc[rb][r] = 0.f;

    for (int sb = 0; sb < 4; ++sb) {   // four 32-token sub-blocks
        const _Float16* Xrow = Xs + (sb * 32) * XSTR;

        // --- logits L(64 x 32) = Qp_h @ Xsub^T + qb ---
        v8f lt[8];
#pragma unroll
        for (int t = 0; t < 8; ++t)
#pragma unroll
            for (int r = 0; r < 8; ++r) lt[t][r] = 0.f;
        for (int kk = 0; kk < ND; kk += 32) {
            const v16h bt0 = load_tile_b(Xrow, 0, kk, XSTR, lane);
            const v16h bt1 = load_tile_b(Xrow, 16, kk, XSTR, lane);
#pragma unroll
            for (int rb = 0; rb < 4; ++rb) {
                const v16h at = load_tile_a(QpH, rb * 16, kk, ND, lane);
                lt[rb * 2 + 0] = wmma16(at, bt0, lt[rb * 2 + 0]);
                lt[rb * 2 + 1] = wmma16(at, bt1, lt[rb * 2 + 1]);
            }
        }

        // --- softmax over the 64 slots of each token column, then /colsum ---
#pragma unroll
        for (int cb = 0; cb < 2; ++cb) {
            float m = -3.0e38f;
#pragma unroll
            for (int rb = 0; rb < 4; ++rb)
#pragma unroll
                for (int r = 0; r < 8; ++r) {
                    const float x = lt[rb * 2 + cb][r] + qbr[rb][r];
                    lt[rb * 2 + cb][r] = x;
                    m = fmaxf(m, x);
                }
            m = fmaxf(m, __shfl_xor(m, 16, 32));   // partner lane holds other 32 slots
            float s = 0.f;
#pragma unroll
            for (int rb = 0; rb < 4; ++rb)
#pragma unroll
                for (int r = 0; r < 8; ++r) {
                    const float e = __expf(lt[rb * 2 + cb][r] - m);
                    lt[rb * 2 + cb][r] = e;
                    s += e;
                }
            s += __shfl_xor(s, 16, 32);
            const float inv = 1.0f / s;
#pragma unroll
            for (int rb = 0; rb < 4; ++rb)
#pragma unroll
                for (int r = 0; r < 8; ++r) {
                    const float w = lt[rb * 2 + cb][r] * inv;
                    lt[rb * 2 + cb][r] = w;
                    cacc[rb][r] += w;              // row-sum over tokens
                }
        }

        // --- spill Wn (64 slots x 32 tokens) row-major f16 into per-wave LDS ---
#pragma unroll
        for (int rb = 0; rb < 4; ++rb)
#pragma unroll
            for (int cb = 0; cb < 2; ++cb)
#pragma unroll
                for (int r = 0; r < 8; ++r)
                    wnS[(rb * 16 + r + 8 * hi) * WSTR + cb * 16 + nn] =
                        (_Float16)lt[rb * 2 + cb][r];

        // --- Vc (32 tokens x 64 d) = Xsub @ Wv_h^T + bv ---
        v8f vt[8];
#pragma unroll
        for (int t = 0; t < 8; ++t)
#pragma unroll
            for (int r = 0; r < 8; ++r) vt[t][r] = 0.f;
        for (int kk = 0; kk < ND; kk += 32) {
            const v16h a0 = load_tile_a(Xrow, 0, kk, XSTR, lane);
            const v16h a1 = load_tile_a(Xrow, 16, kk, XSTR, lane);
#pragma unroll
            for (int cb = 0; cb < 4; ++cb) {
                const v16h bt = load_tile_b(WvH, cb * 16, kk, ND, lane);
                vt[0 * 4 + cb] = wmma16(a0, bt, vt[0 * 4 + cb]);
                vt[1 * 4 + cb] = wmma16(a1, bt, vt[1 * 4 + cb]);
            }
        }
        // store V^T (d-major [64][32]) so it can be reloaded as a B operand
#pragma unroll
        for (int rt = 0; rt < 2; ++rt)
#pragma unroll
            for (int cb = 0; cb < 4; ++cb)
#pragma unroll
                for (int r = 0; r < 8; ++r)
                    vtS[(cb * 16 + nn) * WSTR + rt * 16 + r + 8 * hi] =
                        (_Float16)(vt[rt * 4 + cb][r] + bvr[cb]);
        __syncthreads();

        // --- U += Wn(64x32) @ Vc(32x64): one K=32 WMMA per tile ---
        v16h a3[4], b3[4];
#pragma unroll
        for (int rb = 0; rb < 4; ++rb) a3[rb] = load_tile_a(wnS, rb * 16, 0, WSTR, lane);
#pragma unroll
        for (int cb = 0; cb < 4; ++cb) b3[cb] = load_tile_b(vtS, cb * 16, 0, WSTR, lane);
#pragma unroll
        for (int rb = 0; rb < 4; ++rb)
#pragma unroll
            for (int cb = 0; cb < 4; ++cb)
                uacc[rb * 4 + cb] = wmma16(a3[rb], b3[cb], uacc[rb * 4 + cb]);
        __syncthreads();
    }

    // ---- global accumulation across chunks ----
    float* Ubh = U + ((size_t)(b * NH + h)) * KSL * HD;
#pragma unroll
    for (int rb = 0; rb < 4; ++rb)
#pragma unroll
        for (int cb = 0; cb < 4; ++cb)
#pragma unroll
            for (int r = 0; r < 8; ++r)
                atomicAdd(&Ubh[(rb * 16 + r + 8 * hi) * HD + cb * 16 + nn],
                          uacc[rb * 4 + cb][r]);
#pragma unroll
    for (int rb = 0; rb < 4; ++rb)
#pragma unroll
        for (int r = 0; r < 8; ++r) {
            float v = cacc[rb][r];
            v += __shfl_xor(v, 1, 32);
            v += __shfl_xor(v, 2, 32);
            v += __shfl_xor(v, 4, 32);
            v += __shfl_xor(v, 8, 32);
            if (nn == 0)
                atomicAdd(&Cs[(b * NH + h) * KSL + rb * 16 + r + 8 * hi], v);
        }
}

// ---------------- C: S_hat = U/(C+eps); LayerNorm over D ----------------
__global__ void sse_final(const float* __restrict__ U,
                          const float* __restrict__ Cs,
                          const float* __restrict__ g,
                          const float* __restrict__ bt,
                          float* __restrict__ out) {
    __shared__ float red[256];
    const int b = blockIdx.x >> 6;
    const int k = blockIdx.x & 63;
    float v[2];
#pragma unroll
    for (int j = 0; j < 2; ++j) {
        const int d = threadIdx.x + j * 256;
        const int h = d >> 6, dd = d & 63;
        v[j] = U[((size_t)(b * NH + h) * KSL + k) * HD + dd] /
               (Cs[(b * NH + h) * KSL + k] + EPSW);
    }
    const float mu  = block_sum256(v[0] + v[1], red) * (1.0f / ND);
    const float d0 = v[0] - mu, d1 = v[1] - mu;
    const float var = block_sum256(d0 * d0 + d1 * d1, red) * (1.0f / ND);
    const float rs = rsqrtf(var + LNEPS);
#pragma unroll
    for (int j = 0; j < 2; ++j) {
        const int d = threadIdx.x + j * 256;
        out[((size_t)(b * KSL + k)) * ND + d] = (v[j] - mu) * rs * g[d] + bt[d];
    }
}

extern "C" void kernel_launch(void* const* d_in, const int* in_sizes, int n_in,
                              void* d_out, int out_size, void* d_ws, size_t ws_size,
                              hipStream_t stream) {
    (void)in_sizes; (void)n_in; (void)out_size; (void)ws_size;
    const float* X       = (const float*)d_in[0];
    const float* slots_w = (const float*)d_in[1];
    const float* g_slots = (const float*)d_in[2];
    const float* b_slots = (const float*)d_in[3];
    const float* Wk      = (const float*)d_in[4];
    const float* bk      = (const float*)d_in[5];
    const float* Wv      = (const float*)d_in[6];
    const float* bv      = (const float*)d_in[7];
    const float* g_after = (const float*)d_in[8];
    const float* b_after = (const float*)d_in[9];

    char* ws = (char*)d_ws;
    float*    S    = (float*)(ws + S_OFF);
    _Float16* Wv16 = (_Float16*)(ws + WV_OFF);
    _Float16* Qp   = (_Float16*)(ws + QP_OFF);
    float*    qb   = (float*)(ws + QB_OFF);
    float*    U    = (float*)(ws + U_OFF);
    float*    Cs   = (float*)(ws + CS_OFF);

    // accumulators must be zero every call (harness does not re-poison)
    hipMemsetAsync(ws + U_OFF, 0, U_BYTES + CS_BYTES, stream);

    sse_slot_ln<<<KSL, 256, 0, stream>>>(slots_w, g_slots, b_slots, S);
    sse_cvt16<<<(ND * ND) / 256, 256, 0, stream>>>(Wv, Wv16, ND * ND);
    sse_qproj<<<NH * KSL, 256, 0, stream>>>(S, Wk, bk, Qp, qb);

    const size_t ldsB = (size_t)TC * XSTR * 2           // staged X chunk
                      + (size_t)8 * 2 * KSL * WSTR * 2; // per-wave Wn + V^T
    sse_fused<<<NB * (NT / TC), 256, ldsB, stream>>>(X, Qp, qb, Wv16, bv, U, Cs);

    sse_final<<<NB * KSL, 256, 0, stream>>>(U, Cs, g_after, b_after, (float*)d_out);
}
